// PixelPropagationModule_31645319037131
// MI455X (gfx1250) — matrix operand order repaired and checked by reading, not compiled, not run on hardware
//
#include <hip/hip_runtime.h>
#include <hip/hip_bf16.h>

// ---------------------------------------------------------------------------
// PixelPropagationModule on MI455X (gfx1250):
//   proj (WMMA bf16, LDS-transposed x tiles) ->
//   fused flash-attention (WMMA bf16, online softmax, K stored transposed)
// All WMMA operand fetches are packed b32 (pairs are K-adjacent in VGPRs).
// ---------------------------------------------------------------------------

typedef __attribute__((ext_vector_type(16))) __bf16          v16bf;
typedef __attribute__((ext_vector_type(8)))  float           v8f;
typedef __attribute__((ext_vector_type(8)))  unsigned int    v8u;
typedef __attribute__((ext_vector_type(4)))  unsigned int    v4u;

#define WMMA_BF16(A, B, Cc) \
  __builtin_amdgcn_wmma_f32_16x16x32_bf16(false, (A), false, (B), (short)0, (Cc), false, false)

// Problem constants
constexpr int Bn = 8;       // batch
constexpr int Cc = 256;     // channels
constexpr int Nn = 3136;    // 56*56 pixels
constexpr int Rr = 384;     // stacked projection rows: 64 q + 64 k + 256 v

// Workspace layout (bytes), all 16B aligned
constexpr size_t OFF_WALL = 0;                                    // [384][256] bf16
constexpr size_t OFF_BALL = (size_t)Rr * Cc * 2;                  // [384] f32 (padded)
constexpr size_t OFF_Q    = OFF_BALL + 2048;                      // [B][64][N]  bf16
constexpr size_t OFF_KT   = OFF_Q  + (size_t)Bn * 64 * Nn * 2;    // [B][N][64]  bf16
constexpr size_t OFF_V    = OFF_KT + (size_t)Bn * Nn * 64 * 2;    // [B][256][N] bf16

__device__ __forceinline__ unsigned short f32_to_bf16(float f) {
  unsigned int u = __builtin_bit_cast(unsigned int, f);
  unsigned int r = u + 0x7fffu + ((u >> 16) & 1u);   // round-to-nearest-even
  return (unsigned short)(r >> 16);
}
__device__ __forceinline__ unsigned int pack2bf(float lo, float hi) {
  return (unsigned int)f32_to_bf16(lo) | ((unsigned int)f32_to_bf16(hi) << 16);
}

// ---------------------------------------------------------------------------
__global__ void prep_weights(const float* __restrict__ Wq, const float* __restrict__ bq,
                             const float* __restrict__ Wk, const float* __restrict__ bk,
                             const float* __restrict__ Wv, const float* __restrict__ bv,
                             unsigned short* __restrict__ Wall, float* __restrict__ ball) {
  int idx = blockIdx.x * 256 + threadIdx.x;
  if (idx < Rr * Cc) {
    int r = idx / Cc, c = idx % Cc;
    float w = (r < 64) ? Wq[r * Cc + c]
            : (r < 128) ? Wk[(r - 64) * Cc + c]
                        : Wv[(r - 128) * Cc + c];
    Wall[idx] = f32_to_bf16(w);
    if (idx < Rr)
      ball[idx] = (idx < 64) ? bq[idx] : (idx < 128) ? bk[idx - 64] : bv[idx - 128];
  }
}

// ---------------------------------------------------------------------------
// Projection GEMM:  y[b, r, n] = sum_c Wall[r, c] * x[b, c, n] + ball[r]
// Block: 256 thr (8 waves) -> one 16-pixel column tile, 8 row-tiles (128 rows).
// x tile staged to LDS transposed (f32 -> bf16 inline), shared by all waves.
//   rows   0.. 63 -> q  [b][o][N]
//   rows  64..127 -> kT [b][n][64]   (transposed, packed 16B store per lane)
//   rows 128..383 -> v  [b][c][N]
// ---------------------------------------------------------------------------
__global__ __launch_bounds__(256)
void proj_kernel(const unsigned short* __restrict__ Wall, const float* __restrict__ ball,
                 const float* __restrict__ x,
                 unsigned short* __restrict__ qb, unsigned short* __restrict__ ktb,
                 unsigned short* __restrict__ vb) {
  __shared__ __align__(4) unsigned short xT[16][34];   // [n][c] for current K-step

  const int lane = threadIdx.x & 31;
  const int wid  = threadIdx.x >> 5;
  const int hi   = (lane >> 4) & 1;
  const int ln   = lane & 15;

  const int nt = blockIdx.x % 196;                 // pixel tile (196*16 = 3136)
  const int rg = (blockIdx.x / 196) % 3;           // row group (3*8 row-tiles = 24)
  const int b  = blockIdx.x / (196 * 3);
  const int rt = rg * 8 + wid;                     // this wave's 16-row tile
  const int n0 = nt * 16;

  v8f acc = {};
  for (int c0 = 0; c0 < Cc; c0 += 32) {
    __syncthreads();                               // previous tile fully consumed
    {                                              // stage x[c0:c0+32, n0:n0+16] -> xT[n][c]
      int e = threadIdx.x;                         // 512 elems / 256 thr = 2 each
      int cc0 = e >> 4, nn0 = e & 15;
      xT[nn0][cc0]      = f32_to_bf16(x[(size_t)(b * Cc + c0 + cc0) * Nn + n0 + nn0]);
      int e2 = e + 256;
      int cc1 = e2 >> 4, nn1 = e2 & 15;
      xT[nn1][cc1]      = f32_to_bf16(x[(size_t)(b * Cc + c0 + cc1) * Nn + n0 + nn1]);
    }
    __syncthreads();

    // A fragment: Wall[rt*16 + m, c0 + k] -- packed b32 pairs along k
    v8u au;
#pragma unroll
    for (int v = 0; v < 8; v++) {
      int kb = ((v & 4) << 2) + hi * 8 + 2 * (v & 3);
      au[v] = *(const unsigned int*)(Wall + (rt * 16 + ln) * Cc + c0 + kb);
    }
    v16bf a = __builtin_bit_cast(v16bf, au);

    // B fragment: xT[n = ln][k = table order (8h + s / 16 + 8h + s-8)] -- packed b32 pairs along k (LDS)
    const unsigned int* brow = (const unsigned int*)&xT[ln][0];
    v8u bu;
#pragma unroll
    for (int v = 0; v < 8; v++) bu[v] = brow[((v & 4) << 1) + hi * 4 + (v & 3)];
    v16bf bm = __builtin_bit_cast(v16bf, bu);

    acc = WMMA_BF16(a, bm, acc);
  }

  // Epilogue: bias + scatter to q / kT / v
  float val[8];
#pragma unroll
  for (int r = 0; r < 8; r++) val[r] = acc[r] + ball[rt * 16 + r + 8 * hi];

  if (rt < 4) {                                    // q rows 0..63: [b][o][N]
#pragma unroll
    for (int r = 0; r < 8; r++) {
      int o = rt * 16 + r + 8 * hi;
      qb[(size_t)(b * 64 + o) * Nn + n0 + ln] = f32_to_bf16(val[r]);
    }
  } else if (rt < 8) {                             // k rows: transposed [b][n][64]
    v4u pk;
    pk.x = pack2bf(val[0], val[1]);
    pk.y = pack2bf(val[2], val[3]);
    pk.z = pack2bf(val[4], val[5]);
    pk.w = pack2bf(val[6], val[7]);
    int obase = (rt - 4) * 16 + hi * 8;            // o = obase + r (r-contiguous)
    *(v4u*)(ktb + ((size_t)b * Nn + n0 + ln) * 64 + obase) = pk;
  } else {                                         // v rows: [b][c][N]
#pragma unroll
    for (int r = 0; r < 8; r++) {
      int c = (rt - 8) * 16 + r + 8 * hi;
      vb[(size_t)(b * Cc + c) * Nn + n0 + ln] = f32_to_bf16(val[r]);
    }
  }
}

// ---------------------------------------------------------------------------
// Fused flash attention + epilogue (out = gamma * softmax(QK^T) V^T / l + x).
// Block = 256 thr (8 waves): (batch b, 32 query rows). Per 32-key chunk:
//   waves 0,1: S = Q K^T (packed b32 K loads from kT), online softmax,
//              publish P(bf16) + rescale via LDS.
//   all waves: O[c,i] += V[c, j32] * P^T[j32, i]  (4 WMMAs / wave / chunk)
// ---------------------------------------------------------------------------
__global__ __launch_bounds__(256)
void attn_kernel(const float* __restrict__ x,
                 const unsigned short* __restrict__ qb,
                 const unsigned short* __restrict__ ktb,
                 const unsigned short* __restrict__ vb,
                 const float* __restrict__ gamma_p, float* __restrict__ out) {
  __shared__ __align__(4) unsigned short Qt[32][66];     // [i][o], padded stride
  __shared__ __align__(4) unsigned short Pl[2][16][36];  // [subtile][i][j], padded
  __shared__ float mst[32], lst[32], scl[32];

  const int b  = blockIdx.y;
  const int i0 = blockIdx.x * 32;
  const int tid  = threadIdx.x;
  const int lane = tid & 31;
  const int wid  = tid >> 5;
  const int hi   = (lane >> 4) & 1;
  const int ln   = lane & 15;

  const unsigned short* qbase = qb  + (size_t)b * 64 * Nn;
  const unsigned short* ktbse = ktb + (size_t)b * Nn * 64;
  const unsigned short* vbase = vb  + (size_t)b * Cc * Nn;

  if (tid < 32) { mst[tid] = -1e30f; lst[tid] = 0.f; }
  for (int e = tid; e < 32 * 64; e += 256) {             // stage Q tile
    int o = e >> 5, i = e & 31;
    Qt[i][o] = qbase[(size_t)o * Nn + i0 + i];
  }
  __syncthreads();

  // Q A-fragments: packed b32 pair reads from LDS
  const int si = (wid < 2) ? wid : 0;
  v16bf qa[2];
  {
    const unsigned int* qrow = (const unsigned int*)&Qt[si * 16 + ln][0];
#pragma unroll
    for (int t = 0; t < 2; t++) {
      v8u u;
#pragma unroll
      for (int v = 0; v < 8; v++) {
        int kw = ((v & 4) << 1) + hi * 4 + (v & 3);      // (pair k)/2
        u[v] = qrow[t * 16 + kw];
      }
      qa[t] = __builtin_bit_cast(v16bf, u);
    }
  }

  v8f O[2][2] = {};  // [c-tile][query-subtile]

  for (int jc = 0; jc < 98; jc++) {
    const int j0 = jc * 32;

    // prefetch next chunk's K/V rows into L2 (global_prefetch_b8)
    if (jc + 1 < 98) {
      if (wid < 2)
        __builtin_prefetch(ktbse + (size_t)(j0 + 32 + wid * 16 + ln) * 64, 0, 1);
      else
        __builtin_prefetch(vbase + (size_t)(wid * 32 + ln) * Nn + j0 + 32, 0, 1);
    }

    if (wid < 2) {
      const int s = wid;
      v8f sacc[2] = {};
#pragma unroll
      for (int f = 0; f < 2; f++) {
        // B fragments from kT[b][j][o]: k-adjacent -> packed b32 loads
        const unsigned int* krow =
            (const unsigned int*)(ktbse + (size_t)(j0 + f * 16 + ln) * 64);
#pragma unroll
        for (int t = 0; t < 2; t++) {
          v8u u;
#pragma unroll
          for (int v = 0; v < 8; v++) u[v] = krow[t * 16 + ((v & 4) << 1) + hi * 4 + (v & 3)];
          v16bf kb = __builtin_bit_cast(v16bf, u);
          sacc[f] = WMMA_BF16(qa[t], kb, sacc[f]);
        }
      }
      // Online softmax over this 16x32 score tile
#pragma unroll
      for (int r = 0; r < 8; r++) {
        int row = s * 16 + r + 8 * hi;
        float mx = fmaxf(sacc[0][r], sacc[1][r]);
#pragma unroll
        for (int m = 1; m < 16; m <<= 1) mx = fmaxf(mx, __shfl_xor(mx, m, 32));
        float oldm = mst[row];
        float newm = fmaxf(oldm, mx);
        float sc   = __expf(oldm - newm);
        float p0   = __expf(sacc[0][r] - newm);
        float p1   = __expf(sacc[1][r] - newm);
        float rs   = p0 + p1;
#pragma unroll
        for (int m = 1; m < 16; m <<= 1) rs += __shfl_xor(rs, m, 32);
        if (ln == 0) {
          lst[row] = lst[row] * sc + rs;
          mst[row] = newm;
          scl[row] = sc;
        }
        Pl[s][r + 8 * hi][ln]      = f32_to_bf16(p0);
        Pl[s][r + 8 * hi][16 + ln] = f32_to_bf16(p1);
      }
    }
    __syncthreads();

    // Phase 2: P^T B-fragments from LDS (packed b32, conflict-free stride)
    v16bf pb[2];
#pragma unroll
    for (int s = 0; s < 2; s++) {
      const unsigned int* prow = (const unsigned int*)&Pl[s][ln][0];
      v8u u;
#pragma unroll
      for (int v = 0; v < 8; v++) u[v] = prow[((v & 4) << 1) + hi * 4 + (v & 3)];
      pb[s] = __builtin_bit_cast(v16bf, u);
    }
#pragma unroll
    for (int ct = 0; ct < 2; ct++) {
      const int c = wid * 32 + ct * 16 + ln;               // channel row (A)
      v8u uu;
#pragma unroll
      for (int v = 0; v < 8; v++) {
        int kb = ((v & 4) << 2) + hi * 8 + 2 * (v & 3);    // local j in chunk
        uu[v] = *(const unsigned int*)(vbase + (size_t)c * Nn + j0 + kb);
      }
      v16bf va = __builtin_bit_cast(v16bf, uu);
#pragma unroll
      for (int s = 0; s < 2; s++) {
        float f = scl[s * 16 + ln];                        // rescale (row i = ln)
#pragma unroll
        for (int r = 0; r < 8; r++) O[ct][s][r] *= f;
        O[ct][s] = WMMA_BF16(va, pb[s], O[ct][s]);
      }
    }
    __syncthreads();
  }

  // Epilogue: out = gamma * O / l + x
  const float gamma = gamma_p[0];
#pragma unroll
  for (int ct = 0; ct < 2; ct++) {
#pragma unroll
    for (int s = 0; s < 2; s++) {
      float linv = 1.f / lst[s * 16 + ln];
      int i = i0 + s * 16 + ln;
#pragma unroll
      for (int r = 0; r < 8; r++) {
        int cch = wid * 32 + ct * 16 + r + 8 * hi;
        size_t idx = (size_t)(b * Cc + cch) * Nn + i;
        out[idx] = gamma * (O[ct][s][r] * linv) + x[idx];
      }
    }
  }
}

// ---------------------------------------------------------------------------
extern "C" void kernel_launch(void* const* d_in, const int* in_sizes, int n_in,
                              void* d_out, int out_size, void* d_ws, size_t ws_size,
                              hipStream_t stream) {
  const float* x     = (const float*)d_in[0];
  const float* Wq    = (const float*)d_in[1];
  const float* bq    = (const float*)d_in[2];
  const float* Wk    = (const float*)d_in[3];
  const float* bk    = (const float*)d_in[4];
  const float* Wv    = (const float*)d_in[5];
  const float* bv    = (const float*)d_in[6];
  const float* gamma = (const float*)d_in[7];
  float* out = (float*)d_out;

  char* ws = (char*)d_ws;
  unsigned short* Wall = (unsigned short*)(ws + OFF_WALL);
  float*          ball = (float*)(ws + OFF_BALL);
  unsigned short* qb   = (unsigned short*)(ws + OFF_Q);
  unsigned short* ktb  = (unsigned short*)(ws + OFF_KT);
  unsigned short* vb   = (unsigned short*)(ws + OFF_V);

  prep_weights<<<(Rr * Cc + 255) / 256, 256, 0, stream>>>(Wq, bq, Wk, bk, Wv, bv, Wall, ball);
  // 8 batches * 196 pixel tiles * 3 row groups
  proj_kernel<<<8 * 196 * 3, 256, 0, stream>>>(Wall, ball, x, qb, ktb, vb);
  attn_kernel<<<dim3(98, 8), 256, 0, stream>>>(x, qb, ktb, vb, gamma, out);
}
